// Projector_42142219108895
// MI455X (gfx1250) — matrix operand order repaired and checked by reading, not compiled, not run on hardware
//
#include <hip/hip_runtime.h>
#include <hip/hip_bf16.h>

// ---------------------------------------------------------------------------
// Projector (1x1-conv cross attention), MI455X / gfx1250, wave32 + WMMA bf16.
//   q = Wq@X ; k = Wk@T ; v = Wv@T
//   P = softmax(q^T k / sqrt(C)) ; out = X + v @ P^T
// All GEMMs on v_wmma_f32_16x16x32_bf16 (bf16 in, f32 accumulate).
// Q (f32) staged in d_out; K/V/P (bf16) in d_ws (~21 MB).
// Double-buffered LDS + register-staged global loads + pipelined B-fragments.
// ---------------------------------------------------------------------------

typedef __attribute__((ext_vector_type(16))) __bf16 v16bf;
typedef __attribute__((ext_vector_type(8)))  __bf16 v8bf;
typedef __attribute__((ext_vector_type(8)))  float  v8f;

#define BM   128
#define BN   128
#define BK   32
#define LDT  (BK + 8)      // 40 halves = 80B row stride: 16B-aligned rows, bank spread
#define ATT_BN 256         // full token dim per workgroup in attn kernel

static __device__ inline __bf16 cvt_bf16(float x)  { return (__bf16)x; }
static __device__ inline __bf16 cvt_bf16(__bf16 x) { return x; }

static __device__ inline v16bf join16(v8bf lo, v8bf hi) {
    return __builtin_shufflevector(lo, hi, 0,1,2,3,4,5,6,7,8,9,10,11,12,13,14,15);
}

static __device__ inline v8f wmma_bf16(v16bf a, v16bf b, v8f c) {
    // (neg_a, A, neg_b, B, c_mod, C, reuse_a, reuse_b)
    return __builtin_amdgcn_wmma_f32_16x16x32_bf16(false, a, false, b, (short)0, c,
                                                   false, false);
}

// A frag: lane holds row M, K = half*8 + {0..7} and 16 + half*8 + {0..7}
static __device__ inline v16bf afrag(const __bf16 (*As)[LDT], int row, int half) {
    const __bf16* p = &As[row][half * 8];
    return join16(*(const v8bf*)p, *(const v8bf*)(p + 16));
}
// B frag: lane holds col N, K = half*8 + {0..7} and 16 + half*8 + {0..7}
static __device__ inline v16bf bfrag(const __bf16 (*Bt)[LDT], int col, int half) {
    const __bf16* p = &Bt[col][half * 8];
    return join16(*(const v8bf*)p, *(const v8bf*)(p + 16));
}

// ---------------------------------------------------------------------------
// Generic tiled bf16-WMMA GEMM:  D[b] = A[b](MxK) * B[b](KxN)  (strided views)
// EPI: 0 = store bf16 D, 1 = store f32 D, 2 = store f32 (Xg[idx] + D)
// 256 threads = 8 waves as 4x2; each wave computes a 32x64 sub-tile (2x4 frags).
// Register-staged double-buffered LDS: global loads for tile t+1 issue before
// the compute phase of tile t, LDS stores after it; one barrier per K-step.
// ---------------------------------------------------------------------------
template <typename TA, typename TB, int EPI>
__global__ __launch_bounds__(256)
void gemm_wmma(const TA* __restrict__ Ag, int as0, int as1, long long batchA,
               const TB* __restrict__ Bg, int bs0, int bs1, long long batchB,
               void* __restrict__ Dg, int ds0, long long batchD,
               const float* __restrict__ Xg, int Kdim)
{
    __shared__ alignas(16) __bf16 As[2][BM][LDT];   // [m][k], k contiguous
    __shared__ alignas(16) __bf16 Bt[2][BN][LDT];   // [n][k], k contiguous (transposed)

    const int b    = blockIdx.z;
    const TA* A    = Ag + (size_t)b * batchA;
    const TB* Bm   = Bg + (size_t)b * batchB;
    const int m0   = blockIdx.y * BM;
    const int n0   = blockIdx.x * BN;
    const int tid  = threadIdx.x;
    const int lane = tid & 31;
    const int wid  = tid >> 5;
    const int wr   = wid >> 1;          // 0..3
    const int wc   = wid & 1;           // 0..1
    const int half = lane >> 4;         // which K half this lane holds
    const int lm   = lane & 15;

    // index mapping chosen so consecutive lanes hit stride-1 global addresses
    const bool a_rm = (as1 == 1);
    const bool b_rm = (bs1 == 1);

    TA ra[(BM * BK) / 256];             // 16 staged A elements / thread
    TB rb[(BK * BN) / 256];             // 16 staged B elements / thread

    auto loadA = [&](int k0) {
        #pragma unroll
        for (int j = 0; j < (BM * BK) / 256; j++) {
            int i = tid + j * 256;
            int m = a_rm ? (i >> 5) : (i & 127);
            int k = a_rm ? (i & 31) : (i >> 7);
            ra[j] = A[(size_t)(m0 + m) * as0 + (size_t)(k0 + k) * as1];
        }
    };
    auto storeA = [&](int p) {
        #pragma unroll
        for (int j = 0; j < (BM * BK) / 256; j++) {
            int i = tid + j * 256;
            int m = a_rm ? (i >> 5) : (i & 127);
            int k = a_rm ? (i & 31) : (i >> 7);
            As[p][m][k] = cvt_bf16(ra[j]);
        }
    };
    auto loadB = [&](int k0) {
        #pragma unroll
        for (int j = 0; j < (BK * BN) / 256; j++) {
            int i = tid + j * 256;
            int k = b_rm ? (i >> 7) : (i & 31);
            int n = b_rm ? (i & 127) : (i >> 5);
            rb[j] = Bm[(size_t)(k0 + k) * bs0 + (size_t)(n0 + n) * bs1];
        }
    };
    auto storeB = [&](int p) {
        #pragma unroll
        for (int j = 0; j < (BK * BN) / 256; j++) {
            int i = tid + j * 256;
            int k = b_rm ? (i >> 7) : (i & 31);
            int n = b_rm ? (i & 127) : (i >> 5);
            Bt[p][n][k] = cvt_bf16(rb[j]);
        }
    };

    v8f acc[2][4] = {};

    loadA(0); loadB(0);
    storeA(0); storeB(0);
    __syncthreads();

    const int nk = Kdim / BK;
    for (int t = 0; t < nk; t++) {
        const int p = t & 1;
        if (t + 1 < nk) { loadA((t + 1) * BK); loadB((t + 1) * BK); }
        if (t + 2 < nk) {   // gfx1250 global_prefetch_b8 for the tile after next
            int i = tid;
            int am = a_rm ? (i >> 5) : (i & 127), ak = a_rm ? (i & 31) : (i >> 7);
            int bk = b_rm ? (i >> 7) : (i & 31),  bn = b_rm ? (i & 127) : (i >> 5);
            __builtin_prefetch(&A[(size_t)(m0 + am) * as0 + (size_t)((t + 2) * BK + ak) * as1]);
            __builtin_prefetch(&Bm[(size_t)((t + 2) * BK + bk) * bs0 + (size_t)(n0 + bn) * bs1]);
        }

        v16bf afr[2];
        #pragma unroll
        for (int mi = 0; mi < 2; mi++)
            afr[mi] = afrag(As[p], wr * 32 + mi * 16 + lm, half);

        // software-pipelined B fragments: load ni+1 before consuming ni
        v16bf bcur = bfrag(Bt[p], wc * 64 + lm, half);
        #pragma unroll
        for (int ni = 0; ni < 4; ni++) {
            v16bf bnxt = bcur;
            if (ni < 3) bnxt = bfrag(Bt[p], wc * 64 + (ni + 1) * 16 + lm, half);
            #pragma unroll
            for (int mi = 0; mi < 2; mi++)
                acc[mi][ni] = wmma_bf16(afr[mi], bcur, acc[mi][ni]);
            bcur = bnxt;
        }

        if (t + 1 < nk) { storeA(p ^ 1); storeB(p ^ 1); }
        __syncthreads();
    }

    // Epilogue: D-frag VGPR r -> (row r / r+8 by lane half, col = lane&15)
    const int lrow = half * 8;
    #pragma unroll
    for (int mi = 0; mi < 2; mi++) {
        #pragma unroll
        for (int ni = 0; ni < 4; ni++) {
            const int mbase = m0 + wr * 32 + mi * 16 + lrow;
            const int col   = n0 + wc * 64 + ni * 16 + lm;
            #pragma unroll
            for (int r = 0; r < 8; r++) {
                size_t idx = (size_t)b * batchD + (size_t)(mbase + r) * ds0 + col;
                float val = acc[mi][ni][r];
                if constexpr (EPI == 0)      ((__bf16*)Dg)[idx] = (__bf16)val;
                else if constexpr (EPI == 1) ((float*)Dg)[idx]  = val;
                else                         ((float*)Dg)[idx]  = Xg[idx] + val;
            }
        }
    }
}

// ---------------------------------------------------------------------------
// Fused scores + softmax:  P[b][n][l] = softmax_l( (Q^T K)[n][l] / sqrt(C) )
// Q read as f32 from d_out (element (n,c) at Q[c*N + n]); K is bf16 [C][L].
// 128 pixels x full L=256 per workgroup; wave w owns rows w*16..w*16+15 across
// all 256 columns (16 accumulators) so softmax is wave-local (shfl over halves).
// Ping-pong LDS, one barrier per K-step, pipelined B fragments.
// ---------------------------------------------------------------------------
__global__ __launch_bounds__(256)
void attn_softmax_kernel(const float* __restrict__ Qg, int Npix, long long batchQ,
                         const __bf16* __restrict__ Kg, long long batchK,
                         __bf16* __restrict__ Pg, long long batchP, int Kdim)
{
    __shared__ alignas(16) __bf16 As[2][BM][LDT];       // Q^T tile [pixel][c]
    __shared__ alignas(16) __bf16 Bt[2][ATT_BN][LDT];   // K tile   [l][c]

    const int b    = blockIdx.z;
    const float*  Q  = Qg + (size_t)b * batchQ;
    const __bf16* Kb = Kg + (size_t)b * batchK;
    const int m0   = blockIdx.x * BM;
    const int tid  = threadIdx.x;
    const int lane = tid & 31;
    const int w    = tid >> 5;          // wave 0..7 -> 16-row band
    const int half = lane >> 4;
    const int lm   = lane & 15;

    auto stage = [&](int p, int k0) {
        // A: Q^T view, stride-1 along m (pixels) -> map consecutive tid -> m
        #pragma unroll
        for (int i = tid; i < BM * BK; i += 256) {
            int m = i & 127, k = i >> 7;
            As[p][m][k] = (__bf16)Q[(size_t)(k0 + k) * Npix + (m0 + m)];
        }
        // B: K[c][l] row-major, stride-1 along l -> consecutive tid -> n
        #pragma unroll
        for (int i = tid; i < BK * ATT_BN; i += 256) {
            int k = i >> 8, n = i & 255;
            Bt[p][n][k] = Kb[(size_t)(k0 + k) * 256 + n];
        }
    };

    v8f acc[16] = {};

    stage(0, 0);
    __syncthreads();

    const int nk = Kdim / BK;
    for (int t = 0; t < nk; t++) {
        const int p = t & 1;
        if (t + 1 < nk) stage(p ^ 1, (t + 1) * BK);

        v16bf a = afrag(As[p], w * 16 + lm, half);
        v16bf bcur = bfrag(Bt[p], lm, half);
        #pragma unroll
        for (int ni = 0; ni < 16; ni++) {
            v16bf bnxt = bcur;
            if (ni < 15) bnxt = bfrag(Bt[p], (ni + 1) * 16 + lm, half);
            acc[ni] = wmma_bf16(a, bcur, acc[ni]);
            bcur = bnxt;
        }
        __syncthreads();
    }

    // Row-wise softmax over L=256, entirely in-wave (xor<16 stays in half).
    const float scale = 0.04419417382415922f;   // 1/sqrt(512)
    const int lrow = half * 8;
    #pragma unroll
    for (int r = 0; r < 8; r++) {
        float mx = -3.4e38f;
        #pragma unroll
        for (int ni = 0; ni < 16; ni++) {
            float v = acc[ni][r] * scale;
            acc[ni][r] = v;
            mx = fmaxf(mx, v);
        }
        mx = fmaxf(mx, __shfl_xor(mx, 1, 32));
        mx = fmaxf(mx, __shfl_xor(mx, 2, 32));
        mx = fmaxf(mx, __shfl_xor(mx, 4, 32));
        mx = fmaxf(mx, __shfl_xor(mx, 8, 32));
        float sum = 0.f;
        #pragma unroll
        for (int ni = 0; ni < 16; ni++) {
            float e = __expf(acc[ni][r] - mx);
            acc[ni][r] = e;
            sum += e;
        }
        sum += __shfl_xor(sum, 1, 32);
        sum += __shfl_xor(sum, 2, 32);
        sum += __shfl_xor(sum, 4, 32);
        sum += __shfl_xor(sum, 8, 32);
        const float inv = 1.0f / sum;
        const int n = m0 + w * 16 + lrow + r;    // pixel index
        __bf16* Pr = Pg + (size_t)b * batchP + (size_t)n * 256;
        #pragma unroll
        for (int ni = 0; ni < 16; ni++)
            Pr[ni * 16 + lm] = (__bf16)(acc[ni][r] * inv);
    }
}

// ---------------------------------------------------------------------------
extern "C" void kernel_launch(void* const* d_in, const int* in_sizes, int n_in,
                              void* d_out, int out_size, void* d_ws, size_t ws_size,
                              hipStream_t stream)
{
    (void)in_sizes; (void)n_in; (void)out_size; (void)ws_size;
    const int Bb = 8, C = 512, N = 4096, CT = 512, L = 256;

    const float* X  = (const float*)d_in[0];
    const float* T  = (const float*)d_in[1];
    const float* Wq = (const float*)d_in[2];
    const float* Wk = (const float*)d_in[3];
    const float* Wv = (const float*)d_in[4];
    float* Out = (float*)d_out;

    // workspace: Kbf(2MB) | Vbf(2MB) | Pbf(16.8MB)
    __bf16* Kbf = (__bf16*)d_ws;
    __bf16* Vbf = Kbf + (size_t)Bb * C * L;
    __bf16* Pbf = Vbf + (size_t)Bb * C * L;

    dim3 blk(256);

    // 1) Q = Wq @ X  -> staged as f32 in d_out (overwritten by step 5)
    gemm_wmma<float, float, 1><<<dim3(N / BN, C / BM, Bb), blk, 0, stream>>>(
        Wq, C, 1, 0LL,
        X, N, 1, (long long)C * N,
        Out, N, (long long)C * N, nullptr, C);

    // 2) K = Wk @ T -> bf16
    gemm_wmma<float, float, 0><<<dim3(L / BN, C / BM, Bb), blk, 0, stream>>>(
        Wk, CT, 1, 0LL,
        T, L, 1, (long long)CT * L,
        Kbf, L, (long long)C * L, nullptr, CT);

    // 3) V = Wv @ T -> bf16
    gemm_wmma<float, float, 0><<<dim3(L / BN, C / BM, Bb), blk, 0, stream>>>(
        Wv, CT, 1, 0LL,
        T, L, 1, (long long)CT * L,
        Vbf, L, (long long)C * L, nullptr, CT);

    // 4) P = softmax(Q^T K / sqrt(C)) -> bf16
    attn_softmax_kernel<<<dim3(N / BM, 1, Bb), blk, 0, stream>>>(
        Out, N, (long long)C * N,
        Kbf, (long long)C * L,
        Pbf, (long long)N * L, C);

    // 5) out = X + V @ P^T   (B view of P^T: (k=l, n=pix) strides (1, L))
    gemm_wmma<__bf16, __bf16, 2><<<dim3(N / BN, C / BM, Bb), blk, 0, stream>>>(
        Vbf, L, 1, (long long)C * L,
        Pbf, 1, L, (long long)N * L,
        Out, N, (long long)C * N, X, L);
}
